// Gtu2d_72773925863730
// MI455X (gfx1250) — matrix operand order repaired and checked by reading, not compiled, not run on hardware
//
#include <hip/hip_runtime.h>
#include <hip/hip_bf16.h>

typedef __attribute__((ext_vector_type(16))) _Float16 v16h;
typedef __attribute__((ext_vector_type(8)))  _Float16 v8h;
typedef __attribute__((ext_vector_type(8)))  float    v8f;

#define NN 14
#define MM 14
#define NM 196           // 14*14
#define BATCH 64
#define BNM 12544        // 64*196
#define E_DIM 512
#define D1_DIM 1024
#define PP 729           // 27*27
#define RPE_DIM 512
#define VPITCH 224       // padded v row (halfs), 448B, 16B-aligned

// ---------------------------------------------------------------------------
// Fragment layout (CDNA5 v_wmma_f32_16x16x32_f16, wave32), per ISA 7.12.2:
//   A 16x32 f16: lane L row L%16; half t -> k = (t<8?t:t+8) + 8*(L>=16)
//   B 32x16 f16: lane L col L%16; this code uses half t -> k = t + 16*(L>=16) for A and B
//   C 16x16 f32: lane L col L%16; vgpr r -> m = r + 8*(L>=16)
// LDS A tiles are stored "fragment-ordered": pos(k) = k (contiguous k),
// so a lane's 16 halves are contiguous: base + row*40 + hi*16.  B fragments are
// contiguous in k by construction ([n][k] f16 layout).
// ---------------------------------------------------------------------------
union H2U { _Float16 h[2]; unsigned u; };
__device__ inline unsigned pack2(float a, float b) {
    H2U x; x.h[0] = (_Float16)a; x.h[1] = (_Float16)b; return x.u;
}
__device__ inline v16h cat8(v8h a, v8h b) {
    return __builtin_shufflevector(a, b, 0,1,2,3,4,5,6,7,8,9,10,11,12,13,14,15);
}
__device__ inline v16h frag_lds(const _Float16* base) {           // 2x ds_load_b128
    const v8h* p = (const v8h*)base;
    return cat8(p[0], p[1]);
}
__device__ inline v16h frag_b_global(const _Float16* Bt, int col, int K, int k0, int hi) {
    const v8h* p = (const v8h*)(Bt + (size_t)col * K + k0 + hi * 16);
    return cat8(p[0], p[1]);                                       // 2x global_load_b128
}
__device__ inline void async_wait0() {
#if __has_builtin(__builtin_amdgcn_s_wait_asynccnt)
    __builtin_amdgcn_s_wait_asynccnt(0);
    asm volatile("" ::: "memory");
#else
    asm volatile("s_wait_asynccnt 0x0" ::: "memory");
#endif
}

// ---------------------------------------------------------------------------
// Generic WMMA GEMM:  C = act( (A [⊙ A2t^T]) @ B + bias )
//   A   : M x K f32 row-major;  A2t : K x M f32 or nullptr (fused multiply)
//   Bt  : N x K f16 (pre-transposed weights, read direct from global/L2)
//   mode 0: C f32 row-major MxN; mode 2: C f16 v-layout [n][b][VPITCH]
// Block 256 thr = 8 waves (4m x 2n), tile 128x64, K-chunk 32. N%64==0, K%32==0.
// ---------------------------------------------------------------------------
__global__ __launch_bounds__(256)
void gemm_wmma_kernel(const float* __restrict__ A, const float* __restrict__ A2t,
                      const _Float16* __restrict__ Bt, const float* __restrict__ bias,
                      void* __restrict__ Cout, int M, int N, int K, int act, int mode)
{
    __shared__ _Float16 As16[128 * 40];

    const int tid  = threadIdx.x;
    const int lane = tid & 31;
    const int wave = tid >> 5;
    const int hi   = (lane >> 4) & 1;
    const int wm   = wave >> 1;      // 0..3
    const int wn   = wave & 1;       // 0..1
    const int m0   = blockIdx.x * 128;
    const int n0   = blockIdx.y * 64;

    v8f acc00 = {}, acc01 = {}, acc10 = {}, acc11 = {};

    const int sm_ = tid >> 1;               // staging row
    const int ks_ = (tid & 1) << 4;         // staging k-halve (0 or 16)

    for (int k0 = 0; k0 < K; k0 += 32) {
        // ---- stage A tile (128x32) as f16 fragment-order, 16 elems/thread
        {
            int gm = m0 + sm_;
            float av[16];
            if (gm < M) {
                const float* src = A + (size_t)gm * K + k0 + ks_;
#pragma unroll
                for (int i = 0; i < 16; ++i) av[i] = src[i];
                if (A2t) {
#pragma unroll
                    for (int i = 0; i < 16; ++i)
                        av[i] *= A2t[(size_t)(k0 + ks_ + i) * M + gm];
                }
            } else {
#pragma unroll
                for (int i = 0; i < 16; ++i) av[i] = 0.f;
            }
            // k-blocks: ks_=0 -> pos [0,16) ; ks_=16 -> pos [16,32) (contiguous k)
            unsigned* dst = (unsigned*)(As16 + sm_ * 40);
            int o0 = ks_;
#pragma unroll
            for (int p = 0; p < 4; ++p) dst[(o0 >> 1) + p]        = pack2(av[2*p],   av[2*p+1]);
#pragma unroll
            for (int p = 0; p < 4; ++p) dst[((o0 + 8) >> 1) + p] = pack2(av[8+2*p], av[9+2*p]);
        }
        __syncthreads();

        v16h a0 = frag_lds(As16 + (wm * 32      + (lane & 15)) * 40 + hi * 16);
        v16h a1 = frag_lds(As16 + (wm * 32 + 16 + (lane & 15)) * 40 + hi * 16);
        v16h b0 = frag_b_global(Bt, n0 + wn * 32      + (lane & 15), K, k0, hi);
        v16h b1 = frag_b_global(Bt, n0 + wn * 32 + 16 + (lane & 15), K, k0, hi);
        acc00 = __builtin_amdgcn_wmma_f32_16x16x32_f16(false, a0, false, b0, (short)0, acc00, false, false);
        acc01 = __builtin_amdgcn_wmma_f32_16x16x32_f16(false, a0, false, b1, (short)0, acc01, false, false);
        acc10 = __builtin_amdgcn_wmma_f32_16x16x32_f16(false, a1, false, b0, (short)0, acc10, false, false);
        acc11 = __builtin_amdgcn_wmma_f32_16x16x32_f16(false, a1, false, b1, (short)0, acc11, false, false);
        __syncthreads();
    }

    // ---- epilogue
    v8f accs[4] = {acc00, acc01, acc10, acc11};
#pragma unroll
    for (int sm = 0; sm < 2; ++sm) {
#pragma unroll
        for (int sn = 0; sn < 2; ++sn) {
            v8f acc = accs[sm * 2 + sn];
            int n = n0 + wn * 32 + sn * 16 + (lane & 15);
            float bv = bias ? bias[n] : 0.f;
#pragma unroll
            for (int r = 0; r < 8; ++r) {
                int m = m0 + wm * 32 + sm * 16 + r + hi * 8;
                if (m < M) {
                    float v = acc[r] + bv;
                    if (act == 1) v = v * (1.f / (1.f + __expf(-v)));   // silu
                    if (mode == 2) {
                        int b = m / NM, ij = m - b * NM;
                        ((_Float16*)Cout)[((size_t)n * BATCH + b) * VPITCH + ij] = (_Float16)v;
                    } else {
                        ((float*)Cout)[(size_t)m * N + n] = v;
                    }
                }
            }
        }
    }
}

// ---------------------------------------------------------------------------
// Toeplitz mixing: per channel c, Y_c(196x64) = T_c(196x196) @ V_c(196x64)
// T tile synthesized from coefd[c][di*27+dj] (decay pre-folded, zero for pq>=196).
// V tile copied f16 via global_load_async_to_lds_b128 (rows padded to VPITCH).
// Grid (7 ij-tiles, 1024 channels); 8 waves = 2(ij) x 4(b).
// ---------------------------------------------------------------------------
__global__ __launch_bounds__(256)
void toeplitz_wmma_kernel(const float* __restrict__ coefd,
                          const _Float16* __restrict__ v16,
                          float* __restrict__ y_t)
{
    const int c   = blockIdx.y;
    const int ij0 = blockIdx.x * 32;
    const int tid  = threadIdx.x;
    const int lane = tid & 31;
    const int wave = tid >> 5;
    const int hi   = (lane >> 4) & 1;
    const int wi   = wave >> 2;  // 0..1
    const int wb   = wave & 3;   // 0..3

    __shared__ _Float16 Ts16[32 * 40];
    __shared__ _Float16 Vs16[64 * 40];

    const float*     coef = coefd + (size_t)c * PP;
    const _Float16*  vrow = v16 + (size_t)c * (BATCH * VPITCH);

    v8f acc = {};

    const int ijl = tid >> 3;            // 0..31  (T synthesis row)
    const int pqs = (tid & 7) << 2;      // 0..28  (4 consecutive pq)
    const int vb  = tid >> 2;            // 0..63  (V copy batch)
    const int vps = (tid & 3) << 3;      // 0..24  (8 halfs = 16B)

    for (int k0 = 0; k0 < 224; k0 += 32) {
        // ---- synthesize T tile (32 ij x 32 pq) in fragment order
        {
            int ij = ij0 + ijl;
            int i = ij / MM, j = ij - (ij / MM) * MM;
            float vals[4];
#pragma unroll
            for (int s = 0; s < 4; ++s) {
                int pq = k0 + pqs + s;
                float val = 0.f;
                if (ij < NM && pq < NM) {
                    int p = pq / MM, q = pq - (pq / MM) * MM;
                    val = coef[(i - p + 13) * 27 + (j - q + 13)];
                }
                vals[s] = val;
            }
            int blk = pqs >> 3;
            int pos = pqs;
            unsigned* d = (unsigned*)(Ts16 + ijl * 40 + pos);
            d[0] = pack2(vals[0], vals[1]);
            d[1] = pack2(vals[2], vals[3]);
        }
        // ---- async copy V tile (64 b x 32 pq f16): one b128 per thread
        {
            _Float16* dstp = Vs16 + vb * 40 + vps;
            const _Float16* srcp = vrow + (size_t)vb * VPITCH + k0 + vps;
            unsigned ldsoff = (unsigned)(unsigned long long)dstp;
            asm volatile("global_load_async_to_lds_b128 %0, %1, off"
                         :: "v"(ldsoff), "v"(srcp) : "memory");
        }
        async_wait0();
        __syncthreads();

        v16h a = frag_lds(Ts16 + (wi * 16 + (lane & 15)) * 40 + hi * 16);
        v16h b = frag_lds(Vs16 + (wb * 16 + (lane & 15)) * 40 + hi * 16);
        acc = __builtin_amdgcn_wmma_f32_16x16x32_f16(false, a, false, b, (short)0, acc, false, false);
        __syncthreads();
    }

    int b = wb * 16 + (lane & 15);
    float* dst = y_t + (size_t)c * BNM + (size_t)b * NM;
#pragma unroll
    for (int r = 0; r < 8; ++r) {
        int ij = ij0 + wi * 16 + r + hi * 8;
        if (ij < NM) dst[ij] = acc[r];
    }
}

// ---------------------------------------------------------------------------
// Small kernels
// ---------------------------------------------------------------------------
__global__ void transpose_to_f16_kernel(const float* __restrict__ B,
                                        _Float16* __restrict__ Bt, int K, int N)
{   // Bt[n*K + k] = (f16) B[k*N + n]
    int idx = blockIdx.x * blockDim.x + threadIdx.x;
    if (idx >= K * N) return;
    int n = idx / K, k = idx - (idx / K) * K;
    Bt[idx] = (_Float16)B[(size_t)k * N + n];
}

__global__ void rpe_init_kernel(const float* __restrict__ pw,
                                const float* __restrict__ pb,
                                float* __restrict__ out)
{
    int idx = blockIdx.x * blockDim.x + threadIdx.x;   // 729*512
    if (idx >= PP * RPE_DIM) return;
    int p = idx >> 9, c = idx & 511;
    float ii = (float)(p / 27 - 13);
    float jj = (float)(p % 27 - 13);
    out[idx] = ii * pw[c] + jj * pw[RPE_DIM + c] + pb[c];
}

__global__ __launch_bounds__(256)
void rmsrelu_kernel(const float* __restrict__ in, float* __restrict__ out)
{
    int row = blockIdx.x;
    const float* r = in  + (size_t)row * RPE_DIM;
    float*       o = out + (size_t)row * RPE_DIM;
    int t = threadIdx.x;
    float x0 = r[t], x1 = r[t + 256];
    __shared__ float red[256];
    red[t] = x0 * x0 + x1 * x1;
    __syncthreads();
    for (int off = 128; off > 0; off >>= 1) {
        if (t < off) red[t] += red[t + off];
        __syncthreads();
    }
    float scale = rsqrtf(red[0] * (1.f / (float)RPE_DIM) + 1e-6f);
    float y0 = x0 * scale, y1 = x1 * scale;
    o[t]       = y0 > 0.f ? y0 : 0.f;
    o[t + 256] = y1 > 0.f ? y1 : 0.f;
}

__global__ void decay_transpose_kernel(const float* __restrict__ r,
                                       float* __restrict__ coefd)
{   // coefd[c][p] = r[p][c] * gamma^(|di|+|dj|)
    int idx = blockIdx.x * blockDim.x + threadIdx.x;   // 729*1024
    if (idx >= PP * D1_DIM) return;
    int p = idx >> 10, c = idx & 1023;
    float di = fabsf((float)(p / 27 - 13));
    float dj = fabsf((float)(p % 27 - 13));
    coefd[(size_t)c * PP + p] = r[idx] * __powf(0.999f, di + dj);
}

// ---------------------------------------------------------------------------
extern "C" void kernel_launch(void* const* d_in, const int* in_sizes, int n_in,
                              void* d_out, int out_size, void* d_ws, size_t ws_size,
                              hipStream_t stream) {
    const float* x    = (const float*)d_in[0];
    const float* Wu   = (const float*)d_in[3];
    const float* bu   = (const float*)d_in[4];
    const float* Wv   = (const float*)d_in[5];
    const float* bv   = (const float*)d_in[6];
    const float* Wo   = (const float*)d_in[7];
    const float* bo   = (const float*)d_in[8];
    const float* rpw  = (const float*)d_in[9];
    const float* rpb  = (const float*)d_in[10];
    const float* rw[3] = {(const float*)d_in[11], (const float*)d_in[13], (const float*)d_in[15]};
    const float* rb[3] = {(const float*)d_in[12], (const float*)d_in[14], (const float*)d_in[16]};
    const float* row_w = (const float*)d_in[17];
    const float* row_b = (const float*)d_in[18];
    float* out = (float*)d_out;

    // workspace carve-up
    float* ws    = (float*)d_ws;
    float* u     = ws;                                   // 12544*1024 f32
    float* y_t   = u     + (size_t)BNM * D1_DIM;         // D1 x BNM f32
    float* r_a   = y_t   + (size_t)BNM * D1_DIM;         // 729*512
    float* r_b   = r_a   + (size_t)PP * RPE_DIM;
    float* r_out = r_b   + (size_t)PP * RPE_DIM;         // 729*1024
    float* coefd = r_out + (size_t)PP * D1_DIM;          // 1024*729
    _Float16* v16  = (_Float16*)(coefd + (size_t)D1_DIM * PP);   // [c][b][VPITCH] f16
    _Float16* But  = v16  + (size_t)D1_DIM * BATCH * VPITCH;     // 1024x512
    _Float16* Bvt  = But  + (size_t)D1_DIM * E_DIM;
    _Float16* Wot  = Bvt  + (size_t)D1_DIM * E_DIM;              // 512x1024
    _Float16* w1t  = Wot  + (size_t)E_DIM * D1_DIM;              // 512x512 x3
    _Float16* w2t  = w1t  + (size_t)RPE_DIM * RPE_DIM;
    _Float16* w3t  = w2t  + (size_t)RPE_DIM * RPE_DIM;
    _Float16* rowt = w3t  + (size_t)RPE_DIM * RPE_DIM;           // 1024x512

    // ---- weight pre-transposes to f16 [n][k]
    transpose_to_f16_kernel<<<(E_DIM*D1_DIM+255)/256, 256, 0, stream>>>(Wu, But, E_DIM, D1_DIM);
    transpose_to_f16_kernel<<<(E_DIM*D1_DIM+255)/256, 256, 0, stream>>>(Wv, Bvt, E_DIM, D1_DIM);
    transpose_to_f16_kernel<<<(D1_DIM*E_DIM+255)/256, 256, 0, stream>>>(Wo, Wot, D1_DIM, E_DIM);
    transpose_to_f16_kernel<<<(RPE_DIM*RPE_DIM+255)/256, 256, 0, stream>>>(rw[0], w1t, RPE_DIM, RPE_DIM);
    transpose_to_f16_kernel<<<(RPE_DIM*RPE_DIM+255)/256, 256, 0, stream>>>(rw[1], w2t, RPE_DIM, RPE_DIM);
    transpose_to_f16_kernel<<<(RPE_DIM*RPE_DIM+255)/256, 256, 0, stream>>>(rw[2], w3t, RPE_DIM, RPE_DIM);
    transpose_to_f16_kernel<<<(RPE_DIM*D1_DIM+255)/256, 256, 0, stream>>>(row_w, rowt, RPE_DIM, D1_DIM);

    // ---- RPE MLP -> decayed Toeplitz coefficients (channel-major)
    rpe_init_kernel<<<(PP * RPE_DIM + 255) / 256, 256, 0, stream>>>(rpw, rpb, r_a);
    const _Float16* wts[3] = {w1t, w2t, w3t};
    for (int l = 0; l < 3; ++l) {
        rmsrelu_kernel<<<PP, 256, 0, stream>>>(r_a, r_b);
        gemm_wmma_kernel<<<dim3(6, 8), 256, 0, stream>>>(
            r_b, nullptr, wts[l], rb[l], r_a, PP, RPE_DIM, RPE_DIM, 0, 0);
    }
    rmsrelu_kernel<<<PP, 256, 0, stream>>>(r_a, r_b);
    gemm_wmma_kernel<<<dim3(6, 16), 256, 0, stream>>>(
        r_b, nullptr, rowt, row_b, r_out, PP, D1_DIM, RPE_DIM, 0, 0);
    decay_transpose_kernel<<<(PP * D1_DIM + 255) / 256, 256, 0, stream>>>(r_out, coefd);

    // ---- u = silu(x@Wu+bu) f32 row-major ; v = silu(x@Wv+bv) f16 channel-major
    gemm_wmma_kernel<<<dim3(98, 16), 256, 0, stream>>>(
        x, nullptr, But, bu, u, BNM, D1_DIM, E_DIM, 1, 0);
    gemm_wmma_kernel<<<dim3(98, 16), 256, 0, stream>>>(
        x, nullptr, Bvt, bv, v16, BNM, D1_DIM, E_DIM, 1, 2);

    // ---- y = Toeplitz(v): batched 196x196 @ 196x64 WMMA, async V staging
    toeplitz_wmma_kernel<<<dim3(7, D1_DIM), 256, 0, stream>>>(coefd, v16, y_t);

    // ---- out = (u * y) @ Wo + bo
    gemm_wmma_kernel<<<dim3(98, 8), 256, 0, stream>>>(
        u, y_t, Wot, bo, out, BNM, E_DIM, D1_DIM, 0, 0);
}
